// GatedLinearAttentionARMA_57389353009771
// MI455X (gfx1250) — compile-verified, matrix operand order checked by reading, not run on hardware
//
#include <hip/hip_runtime.h>

#define B_  2
#define L_  512
#define D_  1024
#define H_  16
#define DH_ 64
#define BL_ (B_ * L_)

typedef __attribute__((ext_vector_type(16))) __bf16 v16bf;
typedef __attribute__((ext_vector_type(8)))  __bf16 v8bf;
typedef __attribute__((ext_vector_type(8)))  float  v8f;

// ---------------------------------------------------------------------------
// WMMA helpers (CDNA5 wave32, V_WMMA_F32_16X16X32_BF16)
// ---------------------------------------------------------------------------
__device__ __forceinline__ v8f wmma_bf16(v16bf a, v16bf b, v8f c) {
  // 8 args: (neg_a, A, neg_b, B, c_mod, C, reuse_a, reuse_b)
  return __builtin_amdgcn_wmma_f32_16x16x32_bf16(false, a, false, b, (short)0, c,
                                                 false, false);
}

// ISA 16-bit operand 16x32 layout: lanes 0-15 hold K={0..7,16..23},
// lanes 16-31 hold K={8..15,24..31}; M/N = lane & 15.  For row-major
// (rows x ld) storage each lane's fragment is two contiguous 8-element
// (16-byte) segments -> two 128-bit loads.
__device__ __forceinline__ v16bf frag_ld(const __bf16* base, int r0, int k0,
                                         int ld, int lane) {
  const __bf16* p = base + (r0 + (lane & 15)) * ld + k0 + ((lane & 16) ? 8 : 0);
  v8bf lo = *(const v8bf*)p;
  v8bf hi = *(const v8bf*)(p + 16);
  v16bf r;
#pragma unroll
  for (int e = 0; e < 8; ++e) { r[e] = lo[e]; r[e + 8] = hi[e]; }
  return r;
}

// Transposed bf16 dump of a 16x16 fp32 accumulator tile into dst[n][m]
// (ld elements per row). C layout: VGPR g -> M = g + 8*(lane>=16), N = lane&15,
// so the 8 values per lane are contiguous in m -> one 16-byte store.
__device__ __forceinline__ void dump_frag_T(__bf16* dst, int tile, const v8f& acc,
                                            int lane, int ld) {
  int i = tile >> 2, j = tile & 3;
  int n = 16 * j + (lane & 15);
  int mb = 16 * i + ((lane & 16) ? 8 : 0);
  v8bf v;
#pragma unroll
  for (int g = 0; g < 8; ++g) v[g] = (__bf16)acc[g];
  *(v8bf*)(dst + n * ld + mb) = v;
}

// ---------------------------------------------------------------------------
// Elementwise prep kernels
// ---------------------------------------------------------------------------
__global__ void cast_bf16_kernel(const float* __restrict__ in,
                                 __bf16* __restrict__ out, int n) {
  int i = blockIdx.x * 256 + threadIdx.x;
  if (i < n) out[i] = (__bf16)in[i];
}

// W is K x N row-major (y = x @ W); emit Wt as N x K row-major bf16.
__global__ void transpose_bf16_kernel(const float* __restrict__ in,
                                      __bf16* __restrict__ out, int K, int N) {
  int i = blockIdx.x * 256 + threadIdx.x;
  if (i < K * N) {
    int k = i / N, n = i % N;
    out[n * K + k] = (__bf16)in[i];
  }
}

// ---------------------------------------------------------------------------
// bf16 GEMM, fp32 accumulate: C(MxN,f32) = A(MxK,bf16) @ Bt^T + bias.
// Bt is N x K row-major. 2x2 register blocking: one 32x32 tile per wave
// (4 accumulators, 1 fragment load per WMMA).
// ---------------------------------------------------------------------------
__device__ __forceinline__ void store_tile(float* C, int N, int m0, int n0,
                                           const v8f& acc, const float* bias,
                                           int lane) {
  int n = n0 + (lane & 15);
  int mb = m0 + ((lane & 16) ? 8 : 0);
  float bb = bias ? bias[n] : 0.f;
#pragma unroll
  for (int g = 0; g < 8; ++g) C[(size_t)(mb + g) * N + n] = acc[g] + bb;
}

__global__ __launch_bounds__(128) void gemm_bf16_kernel(
    const __bf16* __restrict__ A, const __bf16* __restrict__ Bt,
    const float* __restrict__ bias, float* __restrict__ C, int M, int N, int K) {
  const int lane = threadIdx.x & 31;
  const int w    = threadIdx.x >> 5;
  const int m0   = (blockIdx.y * 4 + w) * 32;
  const int n0   = blockIdx.x * 32;
  const int koff = (lane & 16) ? 8 : 0;
  const __bf16* ap0 = A  + (size_t)(m0 + (lane & 15)) * K + koff;
  const __bf16* ap1 = ap0 + (size_t)16 * K;
  const __bf16* bp0 = Bt + (size_t)(n0 + (lane & 15)) * K + koff;
  const __bf16* bp1 = bp0 + (size_t)16 * K;
  v8f acc00 = {}, acc01 = {}, acc10 = {}, acc11 = {};
  for (int k0 = 0; k0 < K; k0 += 32) {
    v16bf a0, a1, b0, b1;
    {
      v8bf lo = *(const v8bf*)(ap0 + k0), hi = *(const v8bf*)(ap0 + k0 + 16);
#pragma unroll
      for (int e = 0; e < 8; ++e) { a0[e] = lo[e]; a0[e + 8] = hi[e]; }
    }
    {
      v8bf lo = *(const v8bf*)(ap1 + k0), hi = *(const v8bf*)(ap1 + k0 + 16);
#pragma unroll
      for (int e = 0; e < 8; ++e) { a1[e] = lo[e]; a1[e + 8] = hi[e]; }
    }
    {
      v8bf lo = *(const v8bf*)(bp0 + k0), hi = *(const v8bf*)(bp0 + k0 + 16);
#pragma unroll
      for (int e = 0; e < 8; ++e) { b0[e] = lo[e]; b0[e + 8] = hi[e]; }
    }
    {
      v8bf lo = *(const v8bf*)(bp1 + k0), hi = *(const v8bf*)(bp1 + k0 + 16);
#pragma unroll
      for (int e = 0; e < 8; ++e) { b1[e] = lo[e]; b1[e + 8] = hi[e]; }
    }
    acc00 = wmma_bf16(a0, b0, acc00);
    acc01 = wmma_bf16(a0, b1, acc01);
    acc10 = wmma_bf16(a1, b0, acc10);
    acc11 = wmma_bf16(a1, b1, acc11);
  }
  store_tile(C, N, m0,      n0,      acc00, bias, lane);
  store_tile(C, N, m0,      n0 + 16, acc01, bias, lane);
  store_tile(C, N, m0 + 16, n0,      acc10, bias, lane);
  store_tile(C, N, m0 + 16, n0 + 16, acc11, bias, lane);
}

// ---------------------------------------------------------------------------
// Gate / decay scalars: g = sigmoid(x_head . gw + gw_b), r = silu(K_head . sw + sw_b)
// One wave per (b,l,h).
// ---------------------------------------------------------------------------
__global__ __launch_bounds__(256) void gates_raw_kernel(
    const float* __restrict__ x, const float* __restrict__ Kp,
    const float* __restrict__ gw_w, const float* __restrict__ gw_b,
    const float* __restrict__ sw_w, const float* __restrict__ sw_b,
    float* __restrict__ Graw, float* __restrict__ Rr) {
  int wid  = blockIdx.x * 8 + (threadIdx.x >> 5);
  int lane = threadIdx.x & 31;
  if (wid >= B_ * L_ * H_) return;
  int b = wid / (L_ * H_);
  int l = (wid / H_) % L_;
  int h = wid % H_;
  size_t off = ((size_t)(b * L_ + l)) * D_ + h * DH_;
  float s1 = x[off + lane] * gw_w[lane] + x[off + lane + 32] * gw_w[lane + 32];
  float s2 = Kp[off + lane] * sw_w[lane] + Kp[off + lane + 32] * sw_w[lane + 32];
#pragma unroll
  for (int o = 16; o > 0; o >>= 1) {
    s1 += __shfl_xor(s1, o, 32);
    s2 += __shfl_xor(s2, o, 32);
  }
  if (lane == 0) {
    int idx = (b * H_ + h) * L_ + l;
    Graw[idx] = 1.f / (1.f + expf(-(s1 + gw_b[0])));
    float z = s2 + sw_b[0];
    Rr[idx] = z / (1.f + expf(-z));
  }
}

// 32 sequential log-space cumulative gates (one thread per (b,h) stream).
__global__ void gate_scan_kernel(const float* __restrict__ Graw,
                                 float* __restrict__ Gc) {
  int t = threadIdx.x;
  if (t >= B_ * H_) return;
  const float* g = Graw + t * L_;
  float* gc = Gc + t * L_;
  float acc = 0.f;
  for (int l = 0; l < L_; ++l) {
    acc += logf(fmaxf(g[l], 1e-6f));
    float lc = fminf(30.f, fmaxf(-30.f, acc));
    gc[l] = expf(lc) + 1e-6f;
  }
}

// ---------------------------------------------------------------------------
// Chunked dual-state scan (chunk T=32). One workgroup (8 waves) per (b,h).
// State A  = sum_s k'_s (x) v_s  (k' = K*R/Gc),       O1_t = Gc_t * q_t . A_t
// State S2 = sum_s k2_s (x) e_s, e_s = v_{s+1}-O1_s,  O2_{t+1} = q2_t . S2_t
// States live in WMMA accumulator fragments (2 tiles/wave each).  All LDS
// operands are staged in the layout that makes every fragment read two
// contiguous 16-byte segments.
// ---------------------------------------------------------------------------
__global__ __launch_bounds__(256, 1) void scan_kernel(
    const float* __restrict__ x, const float* __restrict__ Q,
    const float* __restrict__ K1, const float* __restrict__ K2,
    const float* __restrict__ Gc, const float* __restrict__ Rr,
    __bf16* __restrict__ Ybf) {
  const int b = blockIdx.x / H_, h = blockIdx.x % H_;
  const int tid = threadIdx.x, w = tid >> 5, lane = tid & 31;

  // row-major t x d (A-operands)
  __shared__ alignas(16) __bf16 Qb[32 * 64], Q2b[32 * 64];
  __shared__ alignas(16) __bf16 Kpb[32 * 64], K2b[32 * 64];
  // transposed d x t (A-operand of state update / B-operands)
  __shared__ alignas(16) __bf16 Kpt[64 * 32], K2t[64 * 32];
  __shared__ alignas(16) __bf16 Vt[64 * 32], Et[64 * 32];
  // states, stored transposed e x d_k (B-operands)
  __shared__ alignas(16) __bf16 Abt[64 * 64], S2bt[64 * 64];
  // causal score tiles t x s (A-operands)
  __shared__ alignas(16) __bf16 Pb[32 * 32], P2b[32 * 32];
  __shared__ alignas(16) float Vf[33 * 64], O1f[32 * 64], Zf[32 * 64];
  __shared__ float sGc[32], sAl[32], carry[64];

  v8f Af0 = {}, Af1 = {}, Sf0 = {}, Sf1 = {};
  if (tid < 64) carry[tid] = 0.f;

  const size_t rowoff = ((size_t)b * L_) * D_ + h * DH_;
  const float* gcb = Gc + (b * H_ + h) * L_;
  const float* rrb = Rr + (b * H_ + h) * L_;

  for (int t0 = 0; t0 < L_; t0 += 32) {
    // 1. expose current states as bf16 operands (transposed, vectorized store)
    dump_frag_T(Abt, 2 * w, Af0, lane, 64);
    dump_frag_T(Abt, 2 * w + 1, Af1, lane, 64);
    dump_frag_T(S2bt, 2 * w, Sf0, lane, 64);
    dump_frag_T(S2bt, 2 * w + 1, Sf1, lane, 64);
    if (tid < 32) {
      float gc = gcb[t0 + tid];
      sGc[tid] = gc;
      sAl[tid] = rrb[t0 + tid] / gc; // a_t = R_t / Gc_t
    }
    __syncthreads();

    // 2. stage chunk operands + activations (both layouts where needed)
    for (int i = tid; i < 2048; i += 256) {
      int r = i >> 6, c = i & 63;
      size_t g = rowoff + (size_t)(t0 + r) * D_ + c;
      float qf = Q[g];
      Qb[i] = (__bf16)qf;
      float u = -qf * 0.125f;                  // -Q/sqrt(d)
      float lr = (u > 0.f) ? u : 0.02f * u;    // leaky_relu
      Q2b[i] = (__bf16)(-lr);
      float kp = K1[g] * sAl[r];
      Kpb[i] = (__bf16)kp;
      Kpt[c * 32 + r] = (__bf16)kp;
      float k2 = 1.f / (1.f + expf(-K2[g] * 0.000625f)); // 0.02/sqrt(D)
      K2b[i] = (__bf16)k2;
      K2t[c * 32 + r] = (__bf16)k2;
      float vf = x[g];
      Vt[c * 32 + r] = (__bf16)vf;
      Vf[i] = vf;
    }
    if (tid < 64) {
      int l = t0 + 32;
      Vf[32 * 64 + tid] = (l < L_) ? x[rowoff + (size_t)l * D_ + tid] : 0.f;
    }
    __syncthreads();

    // 3. causal score tiles: P = mask(Q @ K'^T), P2 = mask(q2 @ k2^T)
    {
      int ww = w & 3, ti = ww >> 1, tj = ww & 1;
      const __bf16* As = (w < 4) ? Qb : Q2b;
      const __bf16* Bs = (w < 4) ? Kpb : K2b;   // (K'^T)^T == K' row-major
      __bf16* Pd = (w < 4) ? Pb : P2b;
      v8f p = {};
      for (int ks = 0; ks < 64; ks += 32) {
        p = wmma_bf16(frag_ld(As, 16 * ti, ks, 64, lane),
                      frag_ld(Bs, 16 * tj, ks, 64, lane), p);
      }
      int n = 16 * tj + (lane & 15);
      int mb = 16 * ti + ((lane & 16) ? 8 : 0);
#pragma unroll
      for (int g = 0; g < 8; ++g) {
        int m = mb + g;
        Pd[m * 32 + n] = (n <= m) ? (__bf16)p[g] : (__bf16)0.f;
      }
    }
    __syncthreads();

    // 4. O1 = Gc * (Q @ A_prev + P @ V)
    {
      int ti = w >> 2, tj = w & 3;
      v8f o = {};
      for (int ks = 0; ks < 64; ks += 32) {
        o = wmma_bf16(frag_ld(Qb, 16 * ti, ks, 64, lane),
                      frag_ld(Abt, 16 * tj, ks, 64, lane), o);
      }
      o = wmma_bf16(frag_ld(Pb, 16 * ti, 0, 32, lane),
                    frag_ld(Vt, 16 * tj, 0, 32, lane), o);
      int n = 16 * tj + (lane & 15);
      int mb = 16 * ti + ((lane & 16) ? 8 : 0);
#pragma unroll
      for (int g = 0; g < 8; ++g) O1f[(mb + g) * 64 + n] = o[g] * sGc[mb + g];
    }
    __syncthreads();

    // 5. residuals E[t] = V[t+1] - O1[t]  (transposed store)
    for (int i = tid; i < 2048; i += 256) {
      int r = i >> 6, c = i & 63;
      float e = Vf[i + 64] - O1f[i];
      if (t0 + r == L_ - 1) e = 0.f;
      Et[c * 32 + r] = (__bf16)e;
    }
    __syncthreads();

    // 6a. Z = q2 @ S2_prev + P2 @ E   (O2[t+1] = Z[t])
    {
      int ti = w >> 2, tj = w & 3;
      v8f z = {};
      for (int ks = 0; ks < 64; ks += 32) {
        z = wmma_bf16(frag_ld(Q2b, 16 * ti, ks, 64, lane),
                      frag_ld(S2bt, 16 * tj, ks, 64, lane), z);
      }
      z = wmma_bf16(frag_ld(P2b, 16 * ti, 0, 32, lane),
                    frag_ld(Et, 16 * tj, 0, 32, lane), z);
      int n = 16 * tj + (lane & 15);
      int mb = 16 * ti + ((lane & 16) ? 8 : 0);
#pragma unroll
      for (int g = 0; g < 8; ++g) Zf[(mb + g) * 64 + n] = z[g];
    }
    __syncthreads();

    // 6b. Y = O1 + shift(Z); keep Z[31] as cross-chunk carry
    float ncarry = (tid < 64) ? Zf[31 * 64 + tid] : 0.f;
    for (int i = tid; i < 2048; i += 256) {
      int r = i >> 6, c = i & 63;
      float o2 = (r == 0) ? carry[c] : Zf[i - 64];
      Ybf[rowoff + (size_t)(t0 + r) * D_ + c] = (__bf16)(O1f[i] + o2);
    }
    __syncthreads();
    if (tid < 64) carry[tid] = ncarry;

    // 7. state updates: A += K'^T @ V,  S2 += k2^T @ E   (K = 32 = chunk)
    {
      int T0 = 2 * w, T1 = 2 * w + 1;
      int i0 = T0 >> 2, j0 = T0 & 3, i1 = T1 >> 2, j1 = T1 & 3;
      Af0 = wmma_bf16(frag_ld(Kpt, 16 * i0, 0, 32, lane),
                      frag_ld(Vt, 16 * j0, 0, 32, lane), Af0);
      Af1 = wmma_bf16(frag_ld(Kpt, 16 * i1, 0, 32, lane),
                      frag_ld(Vt, 16 * j1, 0, 32, lane), Af1);
      Sf0 = wmma_bf16(frag_ld(K2t, 16 * i0, 0, 32, lane),
                      frag_ld(Et, 16 * j0, 0, 32, lane), Sf0);
      Sf1 = wmma_bf16(frag_ld(K2t, 16 * i1, 0, 32, lane),
                      frag_ld(Et, 16 * j1, 0, 32, lane), Sf1);
    }
    __syncthreads();
  }
}

// ---------------------------------------------------------------------------
// Host launch
// ---------------------------------------------------------------------------
extern "C" void kernel_launch(void* const* d_in, const int* in_sizes, int n_in,
                              void* d_out, int out_size, void* d_ws, size_t ws_size,
                              hipStream_t stream) {
  (void)in_sizes; (void)n_in; (void)out_size; (void)ws_size;
  const float* x    = (const float*)d_in[0];
  const float* q1_w = (const float*)d_in[1];
  const float* q1_b = (const float*)d_in[2];
  const float* k1_w = (const float*)d_in[3];
  const float* k1_b = (const float*)d_in[4];
  const float* k2_w = (const float*)d_in[5];
  const float* k2_b = (const float*)d_in[6];
  const float* gw_w = (const float*)d_in[7];
  const float* gw_b = (const float*)d_in[8];
  const float* sw_w = (const float*)d_in[9];
  const float* sw_b = (const float*)d_in[10];
  const float* cp_w = (const float*)d_in[11];
  const float* cp_b = (const float*)d_in[12];
  float* out = (float*)d_out;

  char* p = (char*)d_ws;
  auto alloc = [&](size_t bytes) -> void* {
    void* r = (void*)p;
    p += (bytes + 255) & ~(size_t)255;
    return r;
  };
  const size_t NXD = (size_t)BL_ * D_; // 1M elements
  __bf16* xbf  = (__bf16*)alloc(NXD * 2);
  __bf16* q1t  = (__bf16*)alloc((size_t)D_ * D_ * 2);
  __bf16* k1t  = (__bf16*)alloc((size_t)D_ * D_ * 2);
  __bf16* k2t  = (__bf16*)alloc((size_t)D_ * D_ * 2);
  __bf16* cpt  = (__bf16*)alloc((size_t)D_ * D_ * 2);
  float*  Qf   = (float*)alloc(NXD * 4);
  float*  K1f  = (float*)alloc(NXD * 4);
  float*  K2f  = (float*)alloc(NXD * 4);
  float*  Graw = (float*)alloc((size_t)B_ * H_ * L_ * 4);
  float*  Gcv  = (float*)alloc((size_t)B_ * H_ * L_ * 4);
  float*  Rr   = (float*)alloc((size_t)B_ * H_ * L_ * 4);
  __bf16* Ybf  = (__bf16*)alloc(NXD * 2);

  // casts & weight transposes
  cast_bf16_kernel<<<(int)(NXD / 256), 256, 0, stream>>>(x, xbf, (int)NXD);
  int wn = D_ * D_;
  transpose_bf16_kernel<<<wn / 256, 256, 0, stream>>>(q1_w, q1t, D_, D_);
  transpose_bf16_kernel<<<wn / 256, 256, 0, stream>>>(k1_w, k1t, D_, D_);
  transpose_bf16_kernel<<<wn / 256, 256, 0, stream>>>(k2_w, k2t, D_, D_);
  transpose_bf16_kernel<<<wn / 256, 256, 0, stream>>>(cp_w, cpt, D_, D_);

  // projections: Q, K, K2 (M=1024, N=1024, K=1024; 32x32 per wave)
  dim3 gg(D_ / 32, BL_ / 128), gb(128);
  gemm_bf16_kernel<<<gg, gb, 0, stream>>>(xbf, q1t, q1_b, Qf, BL_, D_, D_);
  gemm_bf16_kernel<<<gg, gb, 0, stream>>>(xbf, k1t, k1_b, K1f, BL_, D_, D_);
  gemm_bf16_kernel<<<gg, gb, 0, stream>>>(xbf, k2t, k2_b, K2f, BL_, D_, D_);

  // gate / decay scalars, cumulative gate
  gates_raw_kernel<<<(B_ * L_ * H_) / 8, 256, 0, stream>>>(x, K1f, gw_w, gw_b,
                                                           sw_w, sw_b, Graw, Rr);
  gate_scan_kernel<<<1, 32, 0, stream>>>(Graw, Gcv);

  // chunked dual-state scan -> Y (bf16)
  scan_kernel<<<B_ * H_, 256, 0, stream>>>(x, Qf, K1f, K2f, Gcv, Rr, Ybf);

  // output projection
  gemm_bf16_kernel<<<gg, gb, 0, stream>>>(Ybf, cpt, cp_b, out, BL_, D_, D_);
}
